// PerceiverLayer_28183575396422
// MI455X (gfx1250) — matrix operand order repaired and checked by reading, not compiled, not run on hardware
//
#include <hip/hip_runtime.h>
#include <math.h>

// ---------------- problem constants (match reference) ----------------
constexpr int kDIM   = 1024;
constexpr int kHeads = 16;
constexpr int kHD    = 64;     // head dim
constexpr int kFF    = 4096;
constexpr int kB     = 8;
constexpr int kT     = 512;
constexpr int kS     = 4096;

// ---------------- vector types ----------------
typedef __attribute__((ext_vector_type(16))) __bf16 v16bf;
typedef __attribute__((ext_vector_type(8)))  __bf16 v8bf;
typedef __attribute__((ext_vector_type(8)))  float  v8f;
typedef unsigned int       u32;
typedef unsigned long long u64;
typedef __attribute__((ext_vector_type(4))) u32 u32x4;
typedef __attribute__((ext_vector_type(8))) int i32x8;
typedef __attribute__((ext_vector_type(4))) int i32x4;

__device__ __forceinline__ v16bf cat16(v8bf lo, v8bf hi) {
  v16bf a;
#pragma unroll
  for (int i = 0; i < 8; ++i) { a[i] = lo[i]; a[8 + i] = hi[i]; }
  return a;
}

__device__ __forceinline__ v8f wmma_bf16(v16bf a, v16bf b, v8f c) {
  // D(16x16,f32) = A(16x32,bf16) x B(32x16,bf16) + C
  return __builtin_amdgcn_wmma_f32_16x16x32_bf16(false, a, false, b, (short)0, c,
                                                 false, false);
}

__device__ __forceinline__ float gelu_tanh(float x) {
  return 0.5f * x * (1.0f + tanhf(0.7978845608028654f * (x + 0.044715f * x * x * x)));
}

// ---------------- Tensor Data Mover (CDNA5 TDM) ----------------
#if __has_builtin(__builtin_amdgcn_tensor_load_to_lds)
#define HAVE_TDM 1
#else
#define HAVE_TDM 0
#endif

__device__ __forceinline__ void wait_tensorcnt0() {
#if __has_builtin(__builtin_amdgcn_s_wait_tensorcnt)
  __builtin_amdgcn_s_wait_tensorcnt(0);
#else
  asm volatile("s_wait_tensorcnt 0x0" ::: "memory");
#endif
}

#if HAVE_TDM
// Issue a 2D fp32 tile load (tile_w x tile_h, row stride = stride_elems) from
// global memory into LDS at lds_off bytes. D# layout per CDNA5 ISA ch.8:
//   group0: [1:0]=count=1, [63:32]=lds_addr, [120:64]=global_addr, [127:126]=type=2
//   group1: [17:16]=data_size(2=4B), [79:48]=tensor_dim0, [111:80]=tensor_dim1,
//           [127:112]=tile_dim0, [143:128]=tile_dim1, [207:160]=tensor_dim0_stride
__device__ __forceinline__ void tdm_load_2d_f32(const float* gsrc, u32 lds_off,
                                                u32 tensor_w, u32 tensor_h,
                                                u32 tile_w, u32 tile_h,
                                                u64 stride_elems) {
  u64 ga = (u64)(const void*)gsrc;
  u32x4 g0;
  g0.x = 1u;  // count=1, user mode, gather off
  g0.y = lds_off;
  g0.z = (u32)ga;
  g0.w = (u32)((ga >> 32) & 0x01FFFFFFu) | (2u << 30);  // addr[56:32] | type=2

  u64 q0 = ((u64)2u << 16)                          // data_size = 4 bytes
         | ((u64)(tensor_w & 0xFFFFu) << 48);       // tensor_dim0[15:0]
  u64 q1 = ((u64)(tensor_w >> 16) & 0xFFFFu)        // tensor_dim0[31:16]
         | ((u64)tensor_h << 16)                    // tensor_dim1[31:0]
         | ((u64)(tile_w & 0xFFFFu) << 48);         // tile_dim0
  u64 q2 = ((u64)(tile_h & 0xFFFFu))                // tile_dim1 (tile_dim2 = 0)
         | ((stride_elems & 0xFFFFFFFFull) << 32);  // tensor_dim0_stride[31:0]
  u64 q3 = (stride_elems >> 32) & 0xFFFFull;        // tensor_dim0_stride[47:32]

  i32x8 g1;
  g1[0] = (int)(u32)q0; g1[1] = (int)(u32)(q0 >> 32);
  g1[2] = (int)(u32)q1; g1[3] = (int)(u32)(q1 >> 32);
  g1[4] = (int)(u32)q2; g1[5] = (int)(u32)(q2 >> 32);
  g1[6] = (int)(u32)q3; g1[7] = (int)(u32)(q3 >> 32);
  i32x4 z4 = {0, 0, 0, 0};
#if defined(__clang_major__) && (__clang_major__ >= 23)
  i32x8 z8 = {0, 0, 0, 0, 0, 0, 0, 0};
  __builtin_amdgcn_tensor_load_to_lds(g0, g1, z4, z4, z8, 0);
#else
  __builtin_amdgcn_tensor_load_to_lds(g0, g1, z4, z4, 0);
#endif
}
#endif  // HAVE_TDM

// ---------------- generic fused GEMM ----------------
// C[M,N] = epi( A[M,K] @ W[K,N] + bias )
// Block tile 128(M) x 64(N), 256 threads = 8 waves, wave tile 32x32 (2x2 WMMA).
// W tile staged into LDS by the Tensor Data Mover (wave 0), overlapped with the
// A-tile global->bf16 staging; then converted fp32->bf16 transposed for B frags.
enum { EPI_BIAS = 0, EPI_BIAS_RES = 1, EPI_GELU = 2 };

template <typename TA, int EPI, typename TO>
__global__ __launch_bounds__(256) void gemm_kernel(
    const TA* __restrict__ A, const float* __restrict__ W,
    const float* __restrict__ bias, const float* __restrict__ res,
    TO* __restrict__ out, int M, int N, int K) {
  __shared__ alignas(16) __bf16 As[128 * 40];  // 128 rows x 32 K, bf16 (stride 40)
  __shared__ alignas(16) __bf16 Bt[64 * 40];   // 64 N-cols x 32 K, bf16, transposed
#if HAVE_TDM
  __shared__ alignas(16) float  Wst[32 * 64];  // raw fp32 W tile landed by TDM
#endif

  const int tid  = threadIdx.x;
  const int lane = tid & 31;
  const int wv   = tid >> 5;
  const int half = lane >> 4;   // lanes 16..31 -> second K group
  const int ll   = lane & 15;
  const int wm   = (wv & 3) * 32;  // 4 wave-rows
  const int wn   = (wv >> 2) * 32; // 2 wave-cols
  const int m0   = blockIdx.y * 128;
  const int n0   = blockIdx.x * 64;

  v8f acc[2][2] = {};

  for (int k0 = 0; k0 < K; k0 += 32) {
#if HAVE_TDM
    // ---- wave 0: kick off TDM for the W tile (32 x 64 fp32, row stride N) ----
    if (wv == 0) {
      tdm_load_2d_f32(W + (size_t)k0 * N + n0, (u32)(size_t)(void*)Wst,
                      (u32)N, (u32)(K - k0), 64u, 32u, (u64)N);
    }
#endif
    // ---- all threads: stage A tile (128x32) fp32/bf16 -> bf16 LDS ----
#pragma unroll
    for (int it = 0; it < 4; ++it) {
      int idx = tid + it * 256;
      int r = idx >> 3, c = (idx & 7) * 4;
      const TA* ap = A + (size_t)(m0 + r) * K + k0 + c;
#pragma unroll
      for (int u = 0; u < 4; ++u) As[r * 40 + c + u] = (__bf16)(float)ap[u];
    }
    // prefetch next K tile of A (global_prefetch_b8)
    if (k0 + 32 < K) {
      const TA* nap = A + (size_t)(m0 + (tid >> 1)) * K + (k0 + 32) + (tid & 1) * 16;
      __builtin_prefetch((const void*)nap, 0, 1);
    }
#if HAVE_TDM
    if (wv == 0) wait_tensorcnt0();
    __syncthreads();
    // ---- convert TDM-landed fp32 W tile -> bf16 transposed Bt[n][k] ----
#pragma unroll
    for (int it = 0; it < 2; ++it) {
      int idx = tid + it * 256;
      int kr = idx >> 4, c = (idx & 15) * 4;
#pragma unroll
      for (int u = 0; u < 4; ++u) Bt[(c + u) * 40 + kr] = (__bf16)Wst[kr * 64 + c + u];
    }
#else
    // ---- fallback: cooperative global load of W tile, transposed -> Bt ----
#pragma unroll
    for (int it = 0; it < 2; ++it) {
      int idx = tid + it * 256;
      int kr = idx >> 4, c = (idx & 15) * 4;
      const float* wp = W + (size_t)(k0 + kr) * N + n0 + c;
#pragma unroll
      for (int u = 0; u < 4; ++u) Bt[(c + u) * 40 + kr] = (__bf16)wp[u];
    }
#endif
    __syncthreads();

    // ---- fragments (ISA 7.12.2 layouts) ----
    v16bf af[2], bfv[2];
#pragma unroll
    for (int i = 0; i < 2; ++i) {
      const __bf16* rp = As + (wm + i * 16 + ll) * 40;  // A: row = lane&15
      af[i] = cat16(*(const v8bf*)(rp + half * 8),
                    *(const v8bf*)(rp + 16 + half * 8));
    }
#pragma unroll
    for (int j = 0; j < 2; ++j) {
      const __bf16* cp = Bt + (wn + j * 16 + ll) * 40 + half * 8;  // B: col = lane&15
      bfv[j] = cat16(*(const v8bf*)cp, *(const v8bf*)(cp + 16));
    }
#pragma unroll
    for (int i = 0; i < 2; ++i)
#pragma unroll
      for (int j = 0; j < 2; ++j)
        acc[i][j] = wmma_bf16(af[i], bfv[j], acc[i][j]);
    __syncthreads();
  }

  // ---- epilogue: C layout row = j + 8*half (vgpr j), col = lane&15 ----
#pragma unroll
  for (int j = 0; j < 2; ++j) {
    int col = n0 + wn + j * 16 + ll;
    float bval = bias[col];
#pragma unroll
    for (int i = 0; i < 2; ++i) {
#pragma unroll
      for (int vr = 0; vr < 8; ++vr) {
        int row = m0 + wm + i * 16 + vr + half * 8;
        float r = acc[i][j][vr] + bval;
        if constexpr (EPI == EPI_BIAS_RES) r += res[(size_t)row * N + col];
        if constexpr (EPI == EPI_GELU) r = gelu_tanh(r);
        out[(size_t)row * N + col] = (TO)r;
      }
    }
  }
}

// ---------------- flash attention (cross-attention, no mask) ----------------
// One wave per (b, h, 16-row q block). Streams S in blocks of 32.
__global__ __launch_bounds__(32) void attn_kernel(
    const __bf16* __restrict__ q, const __bf16* __restrict__ k,
    const __bf16* __restrict__ v, float* __restrict__ o) {
  __shared__ alignas(16) __bf16 Ps[16 * 40];  // exp'd scores, 16t x 32s
  __shared__ alignas(16) __bf16 Vs[32 * 72];  // V block, 32s x 64d

  const int lane = threadIdx.x;
  const int half = lane >> 4, ll = lane & 15;
  const int t0 = blockIdx.x * 16;
  const int bh = blockIdx.y;
  const int b = bh >> 4, h = bh & 15;
  const size_t qbase  = ((size_t)b * kT + t0) * kDIM + h * kHD;
  const size_t kvbase = ((size_t)b * kS) * kDIM + h * kHD;

  // Q fragments: 16x64 -> two 16x32 A-frags (row = lane&15, d = 8*half+e (e<8), 16+8*half+e-8 (e>=8))
  v16bf aq[2];
  {
    const __bf16* rp = q + qbase + (size_t)ll * kDIM;
#pragma unroll
    for (int s = 0; s < 2; ++s)
      aq[s] = cat16(*(const v8bf*)(rp + s * 32 + half * 8),
                    *(const v8bf*)(rp + s * 32 + 16 + half * 8));
  }

  v8f oacc[4] = {};
  float m[8], l[8];
#pragma unroll
  for (int j = 0; j < 8; ++j) { m[j] = -1e30f; l[j] = 0.f; }

  for (int s0 = 0; s0 < kS; s0 += 32) {
    // ---- scores P[16t x 32s]: two 16x16 tiles, K-frags straight from global ----
    v8f p[2] = {};
#pragma unroll
    for (int c = 0; c < 2; ++c) {
      const __bf16* kp = k + kvbase + (size_t)(s0 + c * 16 + ll) * kDIM;
#pragma unroll
      for (int ds = 0; ds < 2; ++ds) {
        const __bf16* cp = kp + ds * 32 + half * 8;
        v16bf bk = cat16(*(const v8bf*)cp, *(const v8bf*)(cp + 16));
        p[c] = wmma_bf16(aq[ds], bk, p[c]);
      }
    }
    // ---- online softmax; row = j + 8*half, 16-lane halves reduce via shfl_xor ----
    float alpha[8];
#pragma unroll
    for (int j = 0; j < 8; ++j) {
      p[0][j] *= 0.125f;  // 1/sqrt(64)
      p[1][j] *= 0.125f;
      float rm = fmaxf(p[0][j], p[1][j]);
#pragma unroll
      for (int msk = 8; msk >= 1; msk >>= 1) rm = fmaxf(rm, __shfl_xor(rm, msk, 32));
      float mn = fmaxf(m[j], rm);
      alpha[j] = __expf(m[j] - mn);
      m[j] = mn;
      p[0][j] = __expf(p[0][j] - mn);
      p[1][j] = __expf(p[1][j] - mn);
      float rs = p[0][j] + p[1][j];
#pragma unroll
      for (int msk = 8; msk >= 1; msk >>= 1) rs += __shfl_xor(rs, msk, 32);
      l[j] = l[j] * alpha[j] + rs;
    }
#pragma unroll
    for (int n = 0; n < 4; ++n)
#pragma unroll
      for (int j = 0; j < 8; ++j) oacc[n][j] *= alpha[j];

    // ---- P (C-layout) -> LDS -> A-layout frag ----
#pragma unroll
    for (int j = 0; j < 8; ++j) {
      Ps[(j + half * 8) * 40 + ll]      = (__bf16)p[0][j];
      Ps[(j + half * 8) * 40 + 16 + ll] = (__bf16)p[1][j];
    }
    // ---- stage V block [32s x 64d] (coalesced b128 loads) ----
    {
      const __bf16* vp = v + kvbase + (size_t)(s0 + lane) * kDIM;
#pragma unroll
      for (int i = 0; i < 8; ++i)
        *(v8bf*)(Vs + lane * 72 + i * 8) = *(const v8bf*)(vp + i * 8);
    }
    __syncthreads();

    const __bf16* pr = Ps + ll * 40;
    v16bf ap = cat16(*(const v8bf*)(pr + half * 8),
                     *(const v8bf*)(pr + 16 + half * 8));
#pragma unroll
    for (int n = 0; n < 4; ++n) {
      v16bf bv;
#pragma unroll
      for (int i = 0; i < 16; ++i)
        bv[i] = Vs[(half * 8 + i + (i >> 3) * 8) * 72 + n * 16 + ll];  // B: col = lane&15, K = s
      oacc[n] = wmma_bf16(ap, bv, oacc[n]);
    }
    __syncthreads();
  }

  // ---- normalize + store ----
#pragma unroll
  for (int n = 0; n < 4; ++n)
#pragma unroll
    for (int j = 0; j < 8; ++j) {
      int row = t0 + j + half * 8;
      o[((size_t)b * kT + row) * kDIM + h * kHD + n * 16 + ll] = oacc[n][j] / l[j];
    }
}

// ---------------- orchestration ----------------
extern "C" void kernel_launch(void* const* d_in, const int* in_sizes, int n_in,
                              void* d_out, int out_size, void* d_ws, size_t ws_size,
                              hipStream_t stream) {
  (void)in_sizes; (void)n_in; (void)out_size; (void)ws_size;
  const float* x   = (const float*)d_in[0];
  const float* inp = (const float*)d_in[1];
  const float* Wq  = (const float*)d_in[2];
  const float* bq  = (const float*)d_in[3];
  const float* Wk  = (const float*)d_in[4];
  const float* bk  = (const float*)d_in[5];
  const float* Wv  = (const float*)d_in[6];
  const float* bv  = (const float*)d_in[7];
  const float* Wo  = (const float*)d_in[8];
  const float* bo  = (const float*)d_in[9];
  const float* W1  = (const float*)d_in[10];
  const float* b1  = (const float*)d_in[11];
  const float* W2  = (const float*)d_in[12];
  const float* b2  = (const float*)d_in[13];
  float* out = (float*)d_out;

  // workspace layout (bytes), 200 MB total
  char* ws = (char*)d_ws;
  const size_t BT = (size_t)kB * kT, BS = (size_t)kB * kS;
  __bf16* qb = (__bf16*)(ws);                       size_t off = BT * kDIM * 2;   // 8 MB
  __bf16* kb = (__bf16*)(ws + off);                 off += BS * kDIM * 2;          // 64 MB
  __bf16* vb = (__bf16*)(ws + off);                 off += BS * kDIM * 2;          // 64 MB
  float*  at = (float*)(ws + off);                  off += BT * kDIM * 4;          // 16 MB
  float*  y  = (float*)(ws + off);                  off += BT * kDIM * 4;          // 16 MB
  __bf16* hb = (__bf16*)(ws + off);                 off += BT * kFF * 2;           // 32 MB

  dim3 blk(256);
  // Q/K/V projections -> bf16
  gemm_kernel<float, EPI_BIAS, __bf16>
      <<<dim3(kDIM / 64, (int)(BT / 128)), blk, 0, stream>>>(x, Wq, bq, nullptr, qb,
                                                             (int)BT, kDIM, kDIM);
  gemm_kernel<float, EPI_BIAS, __bf16>
      <<<dim3(kDIM / 64, (int)(BS / 128)), blk, 0, stream>>>(inp, Wk, bk, nullptr, kb,
                                                             (int)BS, kDIM, kDIM);
  gemm_kernel<float, EPI_BIAS, __bf16>
      <<<dim3(kDIM / 64, (int)(BS / 128)), blk, 0, stream>>>(inp, Wv, bv, nullptr, vb,
                                                             (int)BS, kDIM, kDIM);
  // flash cross-attention
  attn_kernel<<<dim3(kT / 16, kB * kHeads), dim3(32), 0, stream>>>(qb, kb, vb, at);
  // output projection + residual: y = x + attn @ Wo + bo
  gemm_kernel<float, EPI_BIAS_RES, float>
      <<<dim3(kDIM / 64, (int)(BT / 128)), blk, 0, stream>>>(at, Wo, bo, x, y,
                                                             (int)BT, kDIM, kDIM);
  // MLP: h = gelu(y @ W1 + b1) -> bf16
  gemm_kernel<float, EPI_GELU, __bf16>
      <<<dim3(kFF / 64, (int)(BT / 128)), blk, 0, stream>>>(y, W1, b1, nullptr, hb,
                                                            (int)BT, kFF, kDIM);
  // out = y + h @ W2 + b2
  gemm_kernel<__bf16, EPI_BIAS_RES, float>
      <<<dim3(kDIM / 64, (int)(BT / 128)), blk, 0, stream>>>(hb, W2, b2, y, out,
                                                             (int)BT, kDIM, kFF);
}
